// MaskedLSTMCell_83665962926764
// MI455X (gfx1250) — matrix operand order repaired and checked by reading, not compiled, not run on hardware
//
#include <hip/hip_runtime.h>
#include <hip/hip_bf16.h>

// ---------------------------------------------------------------------------
// Masked LSTM (T=1024, B=64, IN=HID=512) for MI455X / gfx1250.
//
//  * Weights: masked, transposed, f16, PRE-PACKED into the WMMA B lane
//    layout; each wave preloads its 16 k-chunks into 128 VGPRs before the
//    time loop -> zero weight traffic inside the recurrence.
//  * Activations: X and H PRE-PACKED into the WMMA A lane layout; GEMM inner
//    loop = one contiguous 32B v16h global_load per chunk, software-pipelined
//    8 chunks deep with __builtin_amdgcn_sched_barrier(0) fences so the
//    scheduler cannot collapse the pipeline (round-2 asm showed it recycling
//    one A register pair and issuing s_wait_loadcnt 0x0 before every wmma).
//  * 128 persistent WGs (4 batch-tiles x 32 hid-tiles) x 8 waves
//    (4 gates x {X-part, H-part} of K). Partials combined via LDS,
//    sigmoid/tanh via v_exp_f32, cell state in LDS, new H written as one
//    global_store_b128 per lane directly in packed-A layout (double
//    buffered), f32 output stored nontemporal. One device-scope grid
//    barrier per timestep; recurrence-critical waves get raised s_setprio.
// ---------------------------------------------------------------------------

typedef _Float16 f16;
typedef __attribute__((ext_vector_type(16))) _Float16 v16h;
typedef __attribute__((ext_vector_type(8)))  _Float16 v8h;
typedef __attribute__((ext_vector_type(8)))  float    v8f;

#define T_STEPS 1024
#define BATCH   64
#define HID     512
#define INP     512
#define NBLK    128            // 4 m-tiles * 32 h-tiles
#define PITCH   20             // LDS row pitch (floats), 16B-aligned rows

__device__ __forceinline__ float fast_sigmoid(float x) {
    return 1.0f / (1.0f + __expf(-x));
}
__device__ __forceinline__ float fast_tanh(float x) {
    float e = __expf(-2.0f * x);
    return (1.0f - e) / (1.0f + e);
}

// A/B lane layouts (16-bit, 16x16x32):
//  A (16xK per chunk): lanes 0-15 row M=lane, K {0..7,16..23}; lanes 16-31
//    row M=lane-16, K {8..15,24..31}.  slot s -> k = (s<8 ? hs*8+s : 16+hs*8+s-8)
//  B (Kx16 per chunk): lanes 0-15 col N=lane, K {0..7,16..23}; lanes 16-31 K {8..15,24..31}.
//    slot s -> k = (s<8 ? hs*8+s : 16+hs*8+s-8)
// Both packed so each lane's 16 f16 are contiguous (one 32B load).

// --------------------------- prep: weights (B-packed) ----------------------
__global__ void prep_pack_weights(const float* __restrict__ W_x,
                                  const float* __restrict__ W_h,
                                  const float* __restrict__ thr,   // (HID, 8)
                                  f16* __restrict__ Bx, f16* __restrict__ Bh) {
    size_t idx = (size_t)blockIdx.x * blockDim.x + threadIdx.x;
    const size_t PER = (size_t)4 * HID * INP;
    if (idx >= 2 * PER) return;
    bool isH = idx >= PER;
    size_t r = isH ? (idx - PER) : idx;
    // r = ((g*32 + ht)*16 + kc)*512 + lane*16 + s
    int s    = (int)(r         & 15);
    int lane = (int)((r >> 4)  & 31);
    int kc   = (int)((r >> 9)  & 15);
    int ht   = (int)((r >> 13) & 31);
    int g    = (int)((r >> 18) & 3);
    int n = ht * 16 + (lane & 15);                   // output (hidden) index
    int k = kc * 32 + ((s < 8) ? ((lane >> 4) * 8 + s) : (16 + (lane >> 4) * 8 + (s - 8)));  // reduction index
    const float* W = isH ? W_h : W_x;                // W[g][n][k], 512x512
    float w = W[((size_t)g * HID + n) * 512 + k];
    float t = thr[n * 8 + g + (isH ? 4 : 0)];
    float mw = (fabsf(w) - t >= 0.0f) ? w : 0.0f;    // unit_step(|W|-thr)*W
    (isH ? Bh : Bx)[r] = (f16)mw;
}

// --------------------------- prep: X, h0 (A-packed) ------------------------
__global__ void prep_x_h(const float* __restrict__ X, const float* __restrict__ h0,
                         f16* __restrict__ Xp, f16* __restrict__ Hp0) {
    size_t i = (size_t)blockIdx.x * blockDim.x + threadIdx.x;
    const size_t NXP = (size_t)T_STEPS * BATCH * INP;
    if (i < NXP) {
        // i = (((t*4 + m)*16) + kc)*512 + lane*16 + s
        int s    = (int)(i         & 15);
        int lane = (int)((i >> 4)  & 31);
        int kc   = (int)((i >> 9)  & 15);
        int m    = (int)((i >> 13) & 3);
        size_t t = i >> 15;
        int hs = lane >> 4;
        int b  = m * 16 + (lane & 15);
        int k  = kc * 32 + ((s < 8) ? (hs * 8 + s) : (16 + hs * 8 + (s - 8)));
        Xp[i] = (f16)X[(t * BATCH + b) * INP + k];
    } else if (i < NXP + (size_t)BATCH * HID) {
        size_t r = i - NXP;      // (m*16 + kc)*512 + lane*16 + s
        int s    = (int)(r         & 15);
        int lane = (int)((r >> 4)  & 31);
        int kc   = (int)((r >> 9)  & 15);
        int m    = (int)((r >> 13) & 3);
        int hs = lane >> 4;
        int b  = m * 16 + (lane & 15);
        int k  = kc * 32 + ((s < 8) ? (hs * 8 + s) : (16 + hs * 8 + (s - 8)));
        Hp0[r] = (f16)h0[b * HID + k];
    }
}

// --------------------------- prep: sync vars -------------------------------
__global__ void init_sync(unsigned* __restrict__ sync) {
    sync[0] = 0u;   // arrival counter
    sync[1] = 0u;   // epoch flag
}

// --------------------------- grid barrier ----------------------------------
__device__ __forceinline__ void grid_barrier(unsigned* cnt, unsigned* flag,
                                             unsigned nblk, unsigned& epoch) {
    __threadfence();                 // release prior H stores to device scope
    __syncthreads();
    if (threadIdx.x == 0) {
        ++epoch;
        unsigned prev = __hip_atomic_fetch_add(cnt, 1u, __ATOMIC_ACQ_REL,
                                               __HIP_MEMORY_SCOPE_AGENT);
        if (prev == nblk - 1) {
            __hip_atomic_store(cnt, 0u, __ATOMIC_RELAXED, __HIP_MEMORY_SCOPE_AGENT);
            __hip_atomic_store(flag, epoch, __ATOMIC_RELEASE, __HIP_MEMORY_SCOPE_AGENT);
        } else {
            while (__hip_atomic_load(flag, __ATOMIC_ACQUIRE,
                                     __HIP_MEMORY_SCOPE_AGENT) < epoch) {
                __builtin_amdgcn_s_sleep(2);
            }
        }
    }
    __syncthreads();
    __builtin_amdgcn_fence(__ATOMIC_ACQUIRE, "agent");  // invalidate stale H
}

// ---- 512-deep K reduction: 16 wmma, A loads pipelined 8 chunks deep -------
// sched_barrier(0) fences pin the issue order: 8 chunk-loads stay in flight
// behind the wmma chain -> partial s_wait_loadcnt instead of full drains.
__device__ __forceinline__ v8f gemm512(const f16* __restrict__ A,
                                       const v16h* breg) {
    v8f acc = {};
    v16h abuf[8];
#pragma unroll
    for (int i = 0; i < 8; ++i)
        abuf[i] = *(const v16h*)(A + (size_t)i * 512);
    __builtin_amdgcn_sched_barrier(0);
#pragma unroll
    for (int kc = 0; kc < 16; ++kc) {
        acc = __builtin_amdgcn_wmma_f32_16x16x32_f16(
            false, abuf[kc & 7], false, breg[kc], (short)0, acc, false, false);
        if (kc < 8)                      // refill the slot just consumed
            abuf[kc & 7] = *(const v16h*)(A + (size_t)(kc + 8) * 512);
        __builtin_amdgcn_sched_barrier(0);
    }
    return acc;
}

// --------------------------- main persistent kernel ------------------------
__global__ __launch_bounds__(256, 1)
void lstm_persistent(const f16* __restrict__ Xp,
                     const f16* __restrict__ Bx, const f16* __restrict__ Bh,
                     const float* __restrict__ bias, const float* __restrict__ c0,
                     f16* __restrict__ Hp,          // double buffer, contiguous
                     float* __restrict__ out, unsigned* __restrict__ sync) {
    const int tid  = threadIdx.x;
    const int wave = tid >> 5;
    const int lane = tid & 31;
    const int g    = wave & 3;            // gate: 0=i 1=f 2=o 3=c
    const bool isH = wave >= 4;           // H-part of K vs X-part
    const int m    = blockIdx.x & 3;      // batch tile (16 rows)
    const int ht   = blockIdx.x >> 2;     // hidden tile (16 cols)
    const int h0   = ht << 4;
    const int nl   = lane & 15;           // D-layout: column within tile
    const int hs   = lane >> 4;

    __shared__ float lds_part[4][16 * PITCH];
    __shared__ float lds_gate[4][16 * PITCH];
    __shared__ float lds_C[16 * PITCH];

    // raise priority of recurrence-critical waves (H-GEMM + cell update)
    if (isH || wave == 0) __builtin_amdgcn_s_setprio(1);

    // wave-0 row-per-lane assignment for cell update + stores:
    // lane -> row r_ = lane%16, cols ch..ch+7 (ch = 0 or 8)
    const int r_   = lane & 15;
    const int ch   = (lane >> 4) * 8;
    const int kc_h = h0 >> 5;                        // chunk of our h-range
    const int k0   = ((ht & 1) << 4) + ch;           // k within chunk: 0/8/16/24
    const int s0   = (k0 >= 16) ? 8 : 0;             // packed-A slot base
    const int hsp  = (k0 >> 3) & 1;                  // packed-A lane half
    const int hoff = (m * 16 + kc_h) * 512 + (hsp * 16 + r_) * 16 + s0;

    if (wave == 0) {
#pragma unroll
        for (int j = 0; j < 8; ++j)
            lds_C[r_ * PITCH + ch + j] = c0[(m * 16 + r_) * HID + h0 + ch + j];
    }
    const float bv = isH ? 0.0f : bias[g * HID + h0 + nl];

    // preload loop-invariant B (16 chunks x 16 f16 per lane = 128 VGPRs)
    const f16* Bbase = (isH ? Bh : Bx)
                       + ((size_t)(g * 32 + ht) * 16) * 512 + lane * 16;
    v16h breg[16];
#pragma unroll
    for (int kc = 0; kc < 16; ++kc)
        breg[kc] = *(const v16h*)(Bbase + (size_t)kc * 512);

    unsigned epoch = 0;
    __syncthreads();

    for (int t = 0; t < T_STEPS; ++t) {
        v8f acc;
        if (isH) {
            acc = gemm512(Hp + (size_t)(t & 1) * (BATCH * HID)
                             + (m * 16) * 512 + lane * 16, breg);
        } else {
            // warm L2/WGP$ for next step's X tile (past-end at t=1023 is
            // a harmless speculative prefetch into the adjacent ws buffer)
            __builtin_prefetch(Xp + (((size_t)(t + 1) * 4 + m) * 16) * 512
                                  + lane * 16, 0, 1);
            acc = gemm512(Xp + (((size_t)t * 4 + m) * 16) * 512 + lane * 16,
                          breg);
        }

        if (isH) {                         // publish H-part partials
#pragma unroll
            for (int j = 0; j < 8; ++j)
                lds_part[g][(hs * 8 + j) * PITCH + nl] = acc[j];
        }
        __syncthreads();
        if (!isH) {                        // combine + bias + activation
#pragma unroll
            for (int j = 0; j < 8; ++j) {
                int idx = (hs * 8 + j) * PITCH + nl;
                float v = acc[j] + lds_part[g][idx] + bv;
                lds_gate[g][idx] = (g == 3) ? fast_tanh(v) : fast_sigmoid(v);
            }
        }
        __syncthreads();
        if (wave == 0) {                   // cell update + outputs
            f16* Hnext = Hp + (size_t)((t + 1) & 1) * (BATCH * HID);
            float* orow = out + ((size_t)t * BATCH + m * 16 + r_) * HID + h0 + ch;
            v8h hv; v8f ov;
#pragma unroll
            for (int j = 0; j < 8; ++j) {
                int idx = r_ * PITCH + ch + j;
                float I  = lds_gate[0][idx];
                float F  = lds_gate[1][idx];
                float O  = lds_gate[2][idx];
                float Ct = lds_gate[3][idx];
                float C  = F * lds_C[idx] + I * Ct;
                float Hn = O * fast_tanh(C);
                lds_C[idx] = C;
                ov[j] = Hn;
                hv[j] = (f16)Hn;
            }
            __builtin_nontemporal_store(ov, (v8f*)orow);   // streaming output
            *(v8h*)(Hnext + hoff) = hv;    // packed-A layout, one b128 store
        }
        grid_barrier(&sync[0], &sync[1], NBLK, epoch);
    }
}

// --------------------------- launcher --------------------------------------
extern "C" void kernel_launch(void* const* d_in, const int* in_sizes, int n_in,
                              void* d_out, int out_size, void* d_ws, size_t ws_size,
                              hipStream_t stream) {
    (void)in_sizes; (void)n_in; (void)out_size; (void)ws_size;
    const float* X    = (const float*)d_in[0];   // (T,B,IN)
    const float* W_x  = (const float*)d_in[1];   // (4,HID,IN)
    const float* W_h  = (const float*)d_in[2];   // (4,HID,HID)
    const float* bias = (const float*)d_in[3];   // (4,HID)
    const float* thr  = (const float*)d_in[4];   // (HID,8)
    const float* h0   = (const float*)d_in[5];   // (B,HID)
    const float* c0   = (const float*)d_in[6];   // (B,HID)
    float* out = (float*)d_out;                  // (T,B,HID)

    char* ws = (char*)d_ws;
    f16* Bx  = (f16*)(ws);                            //  2 MB packed Wx^T
    f16* Bh  = (f16*)(ws + ((size_t)2  << 20));       //  2 MB packed Wh^T
    f16* Xp  = (f16*)(ws + ((size_t)4  << 20));       // 64 MB packed X
    f16* Hp  = (f16*)(ws + ((size_t)68 << 20));       // 128 KB packed H x2
    unsigned* sync = (unsigned*)(ws + ((size_t)68 << 20) + 2 * 65536);

    {   // mask + pack weights
        size_t n = (size_t)2 * 4 * HID * INP;
        hipLaunchKernelGGL(prep_pack_weights, dim3((n + 255) / 256), dim3(256),
                           0, stream, W_x, W_h, thr, Bx, Bh);
    }
    {   // pack X + h0 to f16 A-layout
        size_t n = (size_t)T_STEPS * BATCH * INP + (size_t)BATCH * HID;
        hipLaunchKernelGGL(prep_x_h, dim3((n + 255) / 256), dim3(256),
                           0, stream, X, h0, Xp, Hp);
    }
    hipLaunchKernelGGL(init_sync, dim3(1), dim3(1), 0, stream, sync);
    hipLaunchKernelGGL(lstm_persistent, dim3(NBLK), dim3(256), 0, stream,
                       Xp, Bx, Bh, bias, c0, Hp, out, sync);
}
